// QoSNet_87789131530409
// MI455X (gfx1250) — matrix operand order repaired and checked by reading, not compiled, not run on hardware
//
#include <hip/hip_runtime.h>

typedef __attribute__((ext_vector_type(16))) _Float16 v16h;
typedef __attribute__((ext_vector_type(8)))  float    v8f;
typedef __attribute__((ext_vector_type(4)))  float    f4;

#define BSZ 8192
#define NL  512
#define TMAX 32
#define DP  128
#define DR  256

// ws layout (float offsets)
#define WS_MAX   0
#define WS_FLOW  16
#define WS_MEAN  (16 + BSZ*DP)
#define WS_RSTD  (WS_MEAN + DP)
#define WS_H16   (WS_RSTD + 112)   /* byte offset 16B-aligned */

#define WMMA(A,B,C) __builtin_amdgcn_wmma_f32_16x16x32_f16(false,(A),false,(B),(short)0,(C),false,false)

#if __has_builtin(__builtin_amdgcn_tanhf)
  #define TANHF(x) __builtin_amdgcn_tanhf(x)
#elif __has_builtin(__builtin_amdgcn_tanh_f32)
  #define TANHF(x) __builtin_amdgcn_tanh_f32(x)
#else
  __device__ __forceinline__ float TANHF(float x) {
      float e = __expf(2.f * x);
      return (e - 1.f) / (e + 1.f);
  }
#endif

__device__ __forceinline__ v16h lds_fragA(const _Float16* rowp, int koff) {
    v16h a;
    *(f4*)&a       = *(const f4*)(rowp + koff);
    *((f4*)&a + 1) = *(const f4*)(rowp + 8 + koff);
    return a;
}
__device__ __forceinline__ v16h lds_fragB(const _Float16* p) {
    v16h b;
    *(f4*)&b       = *(const f4*)p;
    *((f4*)&b + 1) = *(const f4*)(p + 8);
    return b;
}
// sigmoid via native tanh: 1 TRANS op instead of exp+rcp
__device__ __forceinline__ float sigm(float x) { return 0.5f * TANHF(0.5f * x) + 0.5f; }
__device__ __forceinline__ float selu(float x) {
    float s = x > 0.f ? x : 1.6732632423543772f * (__expf(x) - 1.f);
    return 1.0507009873554805f * s;
}

// ---------------- K0: init scratch ----------------
__global__ void k_init(float* ws) {
    if (threadIdx.x == 0 && blockIdx.x == 0) ((unsigned*)ws)[WS_MAX] = 0u;
}

// ---------------- K1: global max of link_capacity ----------------
__global__ void k_max(const float* __restrict__ capa, float* ws) {
    __shared__ float s[256];
    int tid = threadIdx.x;
    float m = 0.f;
    for (int i = blockIdx.x * 256 + tid; i < BSZ * NL; i += gridDim.x * 256)
        m = fmaxf(m, capa[i]);
    s[tid] = m;
    __syncthreads();
    for (int o = 128; o > 0; o >>= 1) {
        if (tid < o) s[tid] = fmaxf(s[tid], s[tid + o]);
        __syncthreads();
    }
    if (tid == 0) atomicMax((unsigned*)&ws[WS_MAX], __float_as_uint(s[0]));
}

// ---------------- K2: transpose readout weights to f16 ----------------
__global__ void k_tr(const float* __restrict__ rW1, const float* __restrict__ rW2, float* ws) {
    _Float16* w1t = (_Float16*)(ws + WS_H16);
    _Float16* w2t = w1t + 3 * DR * DP;
    int idx = blockIdx.x * 256 + threadIdx.x;
    const int T1 = 3 * DP * DR;
    const int T2 = 3 * DR * DR;
    if (idx < T1) {
        int u = idx / (DP * DR), r = idx % (DP * DR);
        int k = r / DR, n = r % DR;
        w1t[u * DR * DP + n * DP + k] = (_Float16)rW1[idx];
    } else if (idx < T1 + T2) {
        int i2 = idx - T1;
        int u = i2 / (DR * DR), r = i2 % (DR * DR);
        int k = r / DR, n = r % DR;
        w2t[u * DR * DR + n * DR + k] = (_Float16)rW2[i2];
    }
}

// ---------------- K3: GRU scan ----------------
#define HSTR 136   // padded LDS row stride in halfs (breaks bank conflicts)
__launch_bounds__(128)
__global__ void k_scan(const float* __restrict__ demand,
                       const float* __restrict__ avail,
                       const float* __restrict__ capa,
                       const float* __restrict__ loss,
                       const int*   __restrict__ path,
                       const int*   __restrict__ hop,
                       const float* __restrict__ wih,
                       const float* __restrict__ whh,
                       const float* __restrict__ bih,
                       const float* __restrict__ bhh,
                       float* ws) {
    __shared__ _Float16 sWhh[384 * HSTR];  // W_hh (n x k) f16, padded
    __shared__ _Float16 sWih8[384 * 8];    // W_ih[:,24:32] (rank-4 input)
    __shared__ float    sBrz[256];
    __shared__ float    sBin[128];
    __shared__ float    sBhn[128];
    __shared__ _Float16 sH[4 * 32 * HSTR]; // per-wave h state, f16
    __shared__ int      sHop[128];

    const int tid  = threadIdx.x;
    const int lane = tid & 31;
    const int w    = tid >> 5;
    const int blockbase = blockIdx.x * 128;
    const float maxc = ws[WS_MAX];
    const float inv_maxc = 1.f / maxc;

    for (int i = tid; i < 384 * 128; i += 128)
        sWhh[(i >> 7) * HSTR + (i & 127)] = (_Float16)whh[i];
    for (int i = tid; i < 384 * 8; i += 128)
        sWih8[i] = (_Float16)wih[(i >> 3) * 32 + 24 + (i & 7)];
    for (int i = tid; i < 256; i += 128) sBrz[i] = bih[i] + bhh[i];
    if (tid < 128) { sBin[tid] = bih[256 + tid]; sBhn[tid] = bhh[256 + tid]; }
    for (int i = tid; i < 4 * 32 * HSTR; i += 128) sH[i] = (_Float16)0.f;
    __syncthreads();
    {
        sHop[tid] = hop[blockbase + tid];
        sH[tid * HSTR + 127] = (_Float16)(demand[blockbase + tid] * inv_maxc);
    }
    __syncthreads();

    _Float16* hW    = sH + w * 32 * HSTR;
    const int* hopW = sHop + w * 32;
    const int nb    = lane & 15;
    const int koffA = (lane < 16) ? 0 : 16;
    const int koffB = (lane < 16) ? 0 : 16;
    const int mhalf = lane >> 4;
    const int rowbase = blockbase + w * 32;

    // register-cache hop values (removes per-step LDS reads of sHop)
    int hopv[2][8];
#pragma unroll
    for (int mt = 0; mt < 2; ++mt)
#pragma unroll
        for (int v = 0; v < 8; ++v) hopv[mt][v] = hopW[mt * 16 + v + 8 * mhalf];
    const int lx = (lane >= 16) ? (lane - 16) : 0;
    int hopx[2];
    hopx[0] = hopW[lx];
    hopx[1] = hopW[16 + lx];
    const int bx0 = rowbase + lx;        // global batch row for x-gather (mt=0)
    const int bx1 = rowbase + 16 + lx;   // (mt=1)

    // zeroed x-fragment template (hoisted)
    v16h AxZ;
#pragma unroll
    for (int e = 0; e < 16; ++e) AxZ[e] = (_Float16)0.f;

#pragma unroll 1
    for (int t = 0; t < TMAX; ++t) {
        // --- h as WMMA A fragments (2 M-tiles x 4 K-chunks) ---
        v16h Ah[2][4];
#pragma unroll
        for (int mt = 0; mt < 2; ++mt) {
            const _Float16* hrow = hW + (mt * 16 + nb) * HSTR;
#pragma unroll
            for (int c = 0; c < 4; ++c) Ah[mt][c] = lds_fragA(hrow + c * 32, koffA);
        }
        // --- sparse x_t A fragments built in registers (K=28..31 only) ---
        v16h Ax[2];
#pragma unroll
        for (int mt = 0; mt < 2; ++mt) {
            v16h a = AxZ;
            if (lane >= 16) {
                int b = mt ? bx1 : bx0;
                if (t < (mt ? hopx[1] : hopx[0])) {
                    int p = path[b * TMAX + t];
                    float av = avail[b * NL + p];
                    float cp = capa[b * NL + p];
                    float ls = loss[b * NL + p];
                    a[12] = (_Float16)(av * inv_maxc);
                    a[13] = (_Float16)(cp * inv_maxc);
                    a[14] = (_Float16)(av / cp);
                    a[15] = (_Float16)ls;
                }
            }
            Ax[mt] = a;
        }
        // --- per 16-col group; unroll 2 so the scheduler interleaves two
        //     independent 30-WMMA groups (hides WMMA->WMMA hazard NOPs) ---
#pragma unroll 2
        for (int j = 0; j < 8; ++j) {
            v8f aR[2], aZ[2], aN[2], aHN[2];
#pragma unroll
            for (int mt = 0; mt < 2; ++mt)
#pragma unroll
                for (int e = 0; e < 8; ++e) { aR[mt][e]=0.f; aZ[mt][e]=0.f; aN[mt][e]=0.f; aHN[mt][e]=0.f; }

            // input contribution (rank-4 via WMMA)
            {
                v16h b;
                f4 q = *(const f4*)(sWih8 + ((j)      * 16 + nb) * 8);
                *(f4*)&b = q; *((f4*)&b + 1) = q;
#pragma unroll
                for (int mt = 0; mt < 2; ++mt) aR[mt] = WMMA(Ax[mt], b, aR[mt]);
                q = *(const f4*)(sWih8 + ((8 + j)  * 16 + nb) * 8);
                *(f4*)&b = q; *((f4*)&b + 1) = q;
#pragma unroll
                for (int mt = 0; mt < 2; ++mt) aZ[mt] = WMMA(Ax[mt], b, aZ[mt]);
                q = *(const f4*)(sWih8 + ((16 + j) * 16 + nb) * 8);
                *(f4*)&b = q; *((f4*)&b + 1) = q;
#pragma unroll
                for (int mt = 0; mt < 2; ++mt) aN[mt] = WMMA(Ax[mt], b, aN[mt]);
            }
            // recurrent contribution: K=128 in 4 chunks, 3 gates, 2 M-tiles
#pragma unroll
            for (int c = 0; c < 4; ++c) {
                v16h b = lds_fragB(sWhh + ((j)      * 16 + nb) * HSTR + c * 32 + koffB);
#pragma unroll
                for (int mt = 0; mt < 2; ++mt) aR[mt]  = WMMA(Ah[mt][c], b, aR[mt]);
                b = lds_fragB(sWhh + ((8 + j)  * 16 + nb) * HSTR + c * 32 + koffB);
#pragma unroll
                for (int mt = 0; mt < 2; ++mt) aZ[mt]  = WMMA(Ah[mt][c], b, aZ[mt]);
                b = lds_fragB(sWhh + ((16 + j) * 16 + nb) * HSTR + c * 32 + koffB);
#pragma unroll
                for (int mt = 0; mt < 2; ++mt) aHN[mt] = WMMA(Ah[mt][c], b, aHN[mt]);
            }
            // gates + state update in C/D layout (lane=col, vgpr=row)
            const float brz_r = sBrz[j * 16 + nb];
            const float brz_z = sBrz[128 + j * 16 + nb];
            const float b_in  = sBin[j * 16 + nb];
            const float b_hn  = sBhn[j * 16 + nb];
#pragma unroll
            for (int mt = 0; mt < 2; ++mt) {
#pragma unroll
                for (int v = 0; v < 8; ++v) {
                    int rloc = mt * 16 + v + 8 * mhalf;
                    float hp = (float)hW[rloc * HSTR + j * 16 + nb];
                    float r  = sigm(aR[mt][v] + brz_r);
                    float z  = sigm(aZ[mt][v] + brz_z);
                    float n  = TANHF(aN[mt][v] + b_in + r * (aHN[mt][v] + b_hn));
                    float hn = (1.f - z) * n + z * hp;
                    hW[rloc * HSTR + j * 16 + nb] = (_Float16)hn;
                    if (hopv[mt][v] == t + 1)
                        ws[WS_FLOW + (rowbase + rloc) * DP + j * 16 + nb] = hn;
                }
            }
        }
    }
}

// ---------------- K4: batchnorm statistics ----------------
__global__ void k_stats(float* ws) {
    __shared__ float s1[256], s2[256];
    int d = blockIdx.x, tid = threadIdx.x;
    float a = 0.f, b = 0.f;
    for (int r = tid; r < BSZ; r += 256) {
        float v = ws[WS_FLOW + r * DP + d];
        a += v; b += v * v;
    }
    s1[tid] = a; s2[tid] = b;
    __syncthreads();
    for (int o = 128; o > 0; o >>= 1) {
        if (tid < o) { s1[tid] += s1[tid + o]; s2[tid] += s2[tid + o]; }
        __syncthreads();
    }
    if (tid == 0) {
        float m = s1[0] / (float)BSZ;
        float var = s2[0] / (float)BSZ - m * m;
        ws[WS_MEAN + d] = m;
        ws[WS_RSTD + d] = rsqrtf(var + 1e-5f);
    }
}

// ---------------- K5: WMMA readout (3 heads, SELU MLP) ----------------
__launch_bounds__(128)
__global__ void k_read(const float* __restrict__ gamma, const float* __restrict__ beta,
                       const float* __restrict__ rb1,   const float* __restrict__ rb2,
                       const float* __restrict__ rW3,   const float* __restrict__ rb3,
                       const float* __restrict__ ws, float* __restrict__ out) {
    __shared__ _Float16 sA[64 * HSTR];
    __shared__ _Float16 sH1[64 * 264];
    __shared__ float sY[4][16];
    const _Float16* w1t = (const _Float16*)(ws + WS_H16);
    const _Float16* w2t = w1t + 3 * DR * DP;

    const int tid = threadIdx.x, lane = tid & 31, w = tid >> 5;
    const int base = blockIdx.x * 64;
    for (int i = tid; i < 64 * DP; i += 128) {
        int r = i >> 7, d = i & 127;
        float v = ws[WS_FLOW + (base + r) * DP + d];
        v = (v - ws[WS_MEAN + d]) * ws[WS_RSTD + d] * gamma[d] + beta[d];
        sA[r * HSTR + d] = (_Float16)v;
    }
    __syncthreads();

    const int nb = lane & 15;
    const int koffA = (lane < 16) ? 0 : 16;
    const int koffB = (lane < 16) ? 0 : 16;
    const int mhalf = lane >> 4;
    const _Float16* aBase  = sA  + (w * 16 + nb) * HSTR;
    const _Float16* a2Base = sH1 + (w * 16 + nb) * 264;

    for (int u = 0; u < 3; ++u) {
#pragma unroll 1
        for (int nt = 0; nt < 16; ++nt) {
            v8f acc;
#pragma unroll
            for (int e = 0; e < 8; ++e) acc[e] = 0.f;
#pragma unroll
            for (int c = 0; c < 4; ++c) {
                v16h a = lds_fragA(aBase + c * 32, koffA);
                const _Float16* bp = w1t + (u * DR + nt * 16 + nb) * DP + c * 32 + koffB;
                v16h b;
                *(f4*)&b = *(const f4*)bp; *((f4*)&b + 1) = *(const f4*)(bp + 8);
                acc = WMMA(a, b, acc);
            }
            float bias = rb1[u * DR + nt * 16 + nb];
#pragma unroll
            for (int v = 0; v < 8; ++v) {
                int m = v + 8 * mhalf;
                sH1[(w * 16 + m) * 264 + nt * 16 + nb] = (_Float16)selu(acc[v] + bias);
            }
        }
        float yacc[8];
#pragma unroll
        for (int v = 0; v < 8; ++v) yacc[v] = 0.f;
#pragma unroll 1
        for (int nt = 0; nt < 16; ++nt) {
            v8f acc;
#pragma unroll
            for (int e = 0; e < 8; ++e) acc[e] = 0.f;
#pragma unroll
            for (int c = 0; c < 8; ++c) {
                v16h a = lds_fragA(a2Base + c * 32, koffA);
                const _Float16* bp = w2t + (u * DR + nt * 16 + nb) * DR + c * 32 + koffB;
                v16h b;
                *(f4*)&b = *(const f4*)bp; *((f4*)&b + 1) = *(const f4*)(bp + 8);
                acc = WMMA(a, b, acc);
            }
            float bias = rb2[u * DR + nt * 16 + nb];
            float w3   = rW3[u * DR + nt * 16 + nb];
#pragma unroll
            for (int v = 0; v < 8; ++v) yacc[v] += selu(acc[v] + bias) * w3;
        }
        __syncthreads();
        if (lane < 16) sY[w][lane] = 0.f;
        __syncthreads();
#pragma unroll
        for (int v = 0; v < 8; ++v) atomicAdd(&sY[w][v + 8 * mhalf], yacc[v]);
        __syncthreads();
        if (lane < 16)
            out[(base + w * 16 + lane) * 3 + u] = sY[w][lane] + rb3[u];
        __syncthreads();
    }
}

extern "C" void kernel_launch(void* const* d_in, const int* in_sizes, int n_in,
                              void* d_out, int out_size, void* d_ws, size_t ws_size,
                              hipStream_t stream) {
    (void)in_sizes; (void)n_in; (void)out_size; (void)ws_size;
    const float* demand = (const float*)d_in[0];
    const float* avail  = (const float*)d_in[1];
    const float* capa   = (const float*)d_in[2];
    const float* loss   = (const float*)d_in[3];
    const int*   path   = (const int*)  d_in[4];
    const int*   hop    = (const int*)  d_in[5];
    const float* wih    = (const float*)d_in[6];
    const float* whh    = (const float*)d_in[7];
    const float* bih    = (const float*)d_in[8];
    const float* bhh    = (const float*)d_in[9];
    const float* gamma  = (const float*)d_in[10];
    const float* beta   = (const float*)d_in[11];
    const float* rW1    = (const float*)d_in[12];
    const float* rb1    = (const float*)d_in[13];
    const float* rW2    = (const float*)d_in[14];
    const float* rb2    = (const float*)d_in[15];
    const float* rW3    = (const float*)d_in[16];
    const float* rb3    = (const float*)d_in[17];
    float* ws  = (float*)d_ws;
    float* out = (float*)d_out;

    k_init <<<1, 1, 0, stream>>>(ws);
    k_max  <<<256, 256, 0, stream>>>(capa, ws);
    k_tr   <<<1152, 256, 0, stream>>>(rW1, rW2, ws);
    k_scan <<<BSZ / 128, 128, 0, stream>>>(demand, avail, capa, loss, path, hop,
                                           wih, whh, bih, bhh, ws);
    k_stats<<<DP, 256, 0, stream>>>(ws);
    k_read <<<BSZ / 64, 128, 0, stream>>>(gamma, beta, rb1, rb2, rW3, rb3, ws, out);
}
